// MultiHeadAttention_46591805227405
// MI455X (gfx1250) — compile-verified, matrix operand order checked by reading, not run on hardware
//
#include <hip/hip_runtime.h>

// ---------------------------------------------------------------------------
// Fused: a1 = (k*sqrt(src+eps)) @ (k*sqrt(dest+eps))^T   [per b,h]
//        a3 = exp2(log2(relu(a1)+eps) * 2/3)
//        out = causal-masked (j>=i) forward cumsum_j(a3)
// B=2 H=16 L=2048 D=64, fp32 in / fp32 out (537 MB output => store-bound;
// matmul runs on v_wmma_f32_16x16x32_f16 to stay off the critical path).
//
// Per workgroup (512 thr / 16 waves): stage the full (k*sqrt(dest+eps)) slab
// for this (b,h) into LDS as f16 (2048 rows x 144B padded stride = 288 KB of
// CDNA5's 320 KB LDS; pad => conflict-free ds_load_b128). Each wave owns a
// 16-row M-tile and scans j serially (cumsum), split into:
//   phase 1 (j<i): row totals only -> shfl_xor butterfly, b128 zero store
//   phase 2 (j==i): full prefix + per-element causal mask
//   phase 3 (j>i): full prefix, unconditional store
// Row-tiles are assigned tile = wave*gridDim.x + blockIdx.x so every WG (and
// every SIMD) gets a uniform mix of cheap (high-i) and expensive (low-i)
// rows: with 288 KB LDS there is exactly 1 WG/WGP, so WG-level balance sets
// the makespan.
// ---------------------------------------------------------------------------

typedef __attribute__((ext_vector_type(16))) _Float16 v16h;
typedef __attribute__((ext_vector_type(8)))  _Float16 v8h;
typedef __attribute__((ext_vector_type(8)))  float    v8f;

constexpr int Bc = 2, Hc = 16, Lc = 2048, Dc = 64;
constexpr int WAVES_PER_WG = 16;
constexpr int GRID_X       = (Lc / 16) / WAVES_PER_WG;  // 8 blocks per (b,h)
constexpr int ROWB         = 144;                    // LDS row stride (128B + 16B pad)
constexpr int LDS_BYTES    = Lc * ROWB;              // 294912 B <= 320 KB

__device__ __forceinline__ unsigned pack2(float a, float b) {
  union { _Float16 h[2]; unsigned u; } x;
  x.h[0] = (_Float16)a; x.h[1] = (_Float16)b;
  return x.u;                                        // -> v_cvt_pk_f16_f32
}

// A fragment for v_wmma_f32_16x16x32_f16 straight from global f32 rows.
// Per-lane K interleave: halves 0..7 = K kb+8*khalf.., halves 8..15 = +16.
__device__ __forceinline__ v16h load_frag_g(const float* __restrict__ row,
                                            int khalf, int kb, float scale) {
  const float4* p0 = reinterpret_cast<const float4*>(row + kb + khalf * 8);
  const float4* p1 = reinterpret_cast<const float4*>(row + kb + 16 + khalf * 8);
  float4 c0 = p0[0], c1 = p0[1], c2 = p1[0], c3 = p1[1];
  v16h v;
  v[0]  = (_Float16)(c0.x * scale); v[1]  = (_Float16)(c0.y * scale);
  v[2]  = (_Float16)(c0.z * scale); v[3]  = (_Float16)(c0.w * scale);
  v[4]  = (_Float16)(c1.x * scale); v[5]  = (_Float16)(c1.y * scale);
  v[6]  = (_Float16)(c1.z * scale); v[7]  = (_Float16)(c1.w * scale);
  v[8]  = (_Float16)(c2.x * scale); v[9]  = (_Float16)(c2.y * scale);
  v[10] = (_Float16)(c2.z * scale); v[11] = (_Float16)(c2.w * scale);
  v[12] = (_Float16)(c3.x * scale); v[13] = (_Float16)(c3.y * scale);
  v[14] = (_Float16)(c3.z * scale); v[15] = (_Float16)(c3.w * scale);
  return v;
}

// B fragment from the pre-scaled f16 LDS slab (ds_load_b128).
__device__ __forceinline__ v16h lds_frag(const char* rowBase, int khalf, int kb) {
  const v8h lo = *reinterpret_cast<const v8h*>(rowBase + (kb + khalf * 8) * 2);
  const v8h hi = *reinterpret_cast<const v8h*>(rowBase + (kb + 16 + khalf * 8) * 2);
  v16h v;
#pragma unroll
  for (int e = 0; e < 8; ++e) { v[e] = lo[e]; v[8 + e] = hi[e]; }
  return v;
}

// One 16x16 Gram tile -> a3 = (relu(a1)+eps)^(2/3), in C layout.
__device__ __forceinline__ void tile_a3(const v16h& a0, const v16h& a1,
                                        const char* sm, int jBase,
                                        int row16, int khalf, float y[8]) {
  const char* rowBase = sm + (jBase + row16) * ROWB;
  v16h b0 = lds_frag(rowBase, khalf, 0);
  v16h b1 = lds_frag(rowBase, khalf, 32);
  v8f c = {0.f, 0.f, 0.f, 0.f, 0.f, 0.f, 0.f, 0.f};
  c = __builtin_amdgcn_wmma_f32_16x16x32_f16(false, a0, false, b0,
                                             (short)0, c, false, false);
  c = __builtin_amdgcn_wmma_f32_16x16x32_f16(false, a1, false, b1,
                                             (short)0, c, false, false);
#pragma unroll
  for (int r = 0; r < 8; ++r) {
    float x = fmaxf(c[r], 0.0f) + 1e-12f;
    y[r] = __builtin_amdgcn_exp2f(__builtin_amdgcn_logf(x) * (2.0f / 3.0f));
  }
}

// Full inclusive prefix over N (16-lane group) + carry update + store.
template <bool MASKED>
__device__ __forceinline__ void scan_store(float y[8], float carry[8],
                                           float* __restrict__ out, size_t outBase,
                                           int iBase, int jBase,
                                           int row16, int khalf) {
#pragma unroll
  for (int r = 0; r < 8; ++r) {
    float t = y[r];
#pragma unroll
    for (int d = 1; d < 16; d <<= 1) {
      float u = __shfl_up(t, (unsigned)d, 16);
      t += (row16 >= d) ? u : 0.0f;
    }
    t += carry[r];
    carry[r] = __shfl(t, 15, 16);                    // row total broadcast
    const int gj = jBase + row16;
    float o = t;
    if (MASKED) {
      const int gi = iBase + r + 8 * khalf;
      o = (gj >= gi) ? t : 0.0f;
    }
    out[outBase + (size_t)(r + 8 * khalf) * Lc + gj] = o;
  }
}

__global__ __launch_bounds__(512)
void affinity_cumsum_kernel(const float* __restrict__ k,
                            const float* __restrict__ src,
                            const float* __restrict__ dest,
                            float* __restrict__ out) {
  extern __shared__ char sm[];
  const int tid   = threadIdx.x;
  const int lane  = tid & 31;
  const int wave  = tid >> 5;
  const int bh    = blockIdx.y;
  // Balanced row-tile assignment: waves of one WG stride across the whole
  // triangle so each WGP gets a uniform phase mix.
  const int tile  = wave * GRID_X + blockIdx.x;      // 0..127, unique
  const int iBase = tile * 16;
  const int row16 = lane & 15;
  const int khalf = lane >> 4;

  const float* kbh    = k    + (size_t)bh * Lc * Dc;
  const float* srcBH  = src  + (size_t)bh * Lc;
  const float* destBH = dest + (size_t)bh * Lc;

  // ---- Stage f16 (k * sqrt(dest+eps)) slab into LDS: 4 rows per thread ----
  for (int r = tid; r < Lc; r += 512) {
    const float4* g = reinterpret_cast<const float4*>(kbh + (size_t)r * Dc);
    const float dj  = sqrtf(destBH[r] + 1e-12f);
    uint4* dst = reinterpret_cast<uint4*>(sm + r * ROWB);
#pragma unroll
    for (int c2 = 0; c2 < 8; ++c2) {
      float4 f0 = g[2 * c2], f1 = g[2 * c2 + 1];
      uint4 q;
      q.x = pack2(f0.x * dj, f0.y * dj);
      q.y = pack2(f0.z * dj, f0.w * dj);
      q.z = pack2(f1.x * dj, f1.y * dj);
      q.w = pack2(f1.z * dj, f1.w * dj);
      dst[c2] = q;                                   // ds_store_b128
    }
  }
  __syncthreads();

  // ---- A fragments (rows iBase..iBase+15, scaled by sqrt(src+eps)) ----
  const int   ai    = iBase + row16;
  const float sA    = sqrtf(srcBH[ai] + 1e-12f);
  const float* arow = kbh + (size_t)ai * Dc;
  const v16h a0 = load_frag_g(arow, khalf, 0,  sA);
  const v16h a1 = load_frag_g(arow, khalf, 32, sA);

  float carry[8];
#pragma unroll
  for (int r = 0; r < 8; ++r) carry[r] = 0.0f;

  const size_t outBase = ((size_t)bh * Lc + iBase) * Lc;
  float y[8];
  int jBase = 0;

  // ---- Phase 1: fully masked tiles (j < i). Row totals only; store zeros.
  for (; jBase < iBase; jBase += 16) {
    tile_a3(a0, a1, sm, jBase, row16, khalf, y);
#pragma unroll
    for (int r = 0; r < 8; ++r) {
      float t = y[r];
#pragma unroll
      for (int d = 1; d < 16; d <<= 1) t += __shfl_xor(t, d);  // 16-group sum
      carry[r] += t;
    }
    // zero-fill the 16x16 tile with 2x b128 per lane (row = lane>>1)
    float4 z = {0.f, 0.f, 0.f, 0.f};
    float4* zp = reinterpret_cast<float4*>(
        out + outBase + (size_t)(lane >> 1) * Lc + jBase + (lane & 1) * 8);
    zp[0] = z; zp[1] = z;
  }

  // ---- Phase 2: diagonal tile (j == i), per-element causal mask.
  tile_a3(a0, a1, sm, jBase, row16, khalf, y);
  scan_store<true>(y, carry, out, outBase, iBase, jBase, row16, khalf);
  jBase += 16;

  // ---- Phase 3: fully visible tiles (j > i), unconditional store.
  for (; jBase < Lc; jBase += 16) {
    tile_a3(a0, a1, sm, jBase, row16, khalf, y);
    scan_store<false>(y, carry, out, outBase, iBase, jBase, row16, khalf);
  }
}

extern "C" void kernel_launch(void* const* d_in, const int* in_sizes, int n_in,
                              void* d_out, int out_size, void* d_ws, size_t ws_size,
                              hipStream_t stream) {
  const float* k    = (const float*)d_in[0];
  const float* src  = (const float*)d_in[1];
  const float* dest = (const float*)d_in[2];
  float* out = (float*)d_out;
  (void)in_sizes; (void)n_in; (void)out_size; (void)d_ws; (void)ws_size;

  dim3 grid(GRID_X, Bc * Hc);                        // (8, 32)
  dim3 block(512);                                   // 16 waves
  hipLaunchKernelGGL(affinity_cumsum_kernel, grid, block, LDS_BYTES, stream,
                     k, src, dest, out);
}
